// SequenceLabelingLSTM_CRF_35845797052719
// MI455X (gfx1250) — matrix operand order repaired and checked by reading, not compiled, not run on hardware
//
#include <hip/hip_runtime.h>
#include <hip/hip_bf16.h>

typedef __attribute__((ext_vector_type(16))) _Float16 v16h;
typedef __attribute__((ext_vector_type(8)))  _Float16 v8h;
typedef __attribute__((ext_vector_type(8)))  float    v8f;

#define BB 64      // batch
#define TT 512     // time
#define II 512     // input dim
#define HH 1024    // hidden
#define GG 4096    // 4*H (gates)
#define NC 20      // classes
#define CP 32      // padded classes for WMMA N

// ---------------------------------------------------------------------------
// small utility kernels
// ---------------------------------------------------------------------------
__global__ void cast_f32_to_f16(const float* __restrict__ src,
                                _Float16* __restrict__ dst, long n) {
  long i = (long)blockIdx.x * blockDim.x + threadIdx.x;
  if (i < n) dst[i] = (_Float16)src[i];
}

__global__ void add_bias_pair(const float* __restrict__ a,
                              const float* __restrict__ b,
                              float* __restrict__ out, int n) {
  int i = blockIdx.x * blockDim.x + threadIdx.x;
  if (i < n) out[i] = a[i] + b[i];
}

// pad fc_w (20x1024) -> f16 (32x1024, zero rows 20..31), fc_b -> fp32[32]
__global__ void pad_fc(const float* __restrict__ fcw, const float* __restrict__ fcb,
                       _Float16* __restrict__ w16, float* __restrict__ b32) {
  int i = blockIdx.x * blockDim.x + threadIdx.x;
  if (i < CP * HH) {
    int r = i / HH, c = i % HH;
    w16[i] = (_Float16)((r < NC) ? fcw[r * HH + c] : 0.0f);
  }
  if (i < CP) b32[i] = (i < NC) ? fcb[i] : 0.0f;
}

__global__ void zero_state(_Float16* __restrict__ h, float* __restrict__ c, int n) {
  int i = blockIdx.x * blockDim.x + threadIdx.x;
  if (i < n) { h[i] = (_Float16)0.0f; c[i] = 0.0f; }
}

__global__ void zero_scalar(float* p) {
  if (blockIdx.x == 0 && threadIdx.x == 0) p[0] = 0.0f;
}

// ---------------------------------------------------------------------------
// Branch-free WMMA GEMM:  Out[M x N] = A[M x K] * Bm[N x K]^T (+bias)(+addend)
// Each wave computes a 16(M) x TN*16(N) strip: one A fragment shared by TN
// accumulators.  Waves in a block fan out along N:
//   mTile = blockIdx.y*16
//   nBase = (blockIdx.x*wavesPerBlock + wave) * TN*16
// Grid must tile N exactly: gridDim.x * wavesPerBlock * TN*16 == N.
// K multiple of 32, M multiple of 16.  No bounds checks -> EXEC stays full,
// no execz branches around the WMMAs, loads clause/pipeline freely.
// ---------------------------------------------------------------------------
__device__ __forceinline__ v16h frag_a(const _Float16* p) {
  // A 16x32 f16 layout: lanes 0-15 hold K {0..7,16..23}; lanes 16-31 {8..15,24..31}
  v8h lo = *(const v8h*)(p);
  v8h hi = *(const v8h*)(p + 16);
  return __builtin_shufflevector(lo, hi, 0,1,2,3,4,5,6,7,8,9,10,11,12,13,14,15);
}

template <int TN, bool HAS_BIAS, bool HAS_ADD>
__global__ __launch_bounds__(256) void gemm_f16_wmma(
    const _Float16* __restrict__ A, int lda,
    const _Float16* __restrict__ Bm, int ldb,
    const float* __restrict__ bias,
    const float* __restrict__ addend, long addStride,
    float* __restrict__ Out, int ldc, int K)
{
  const int lane = threadIdx.x & 31;
  const int wave = threadIdx.x >> 5;
  const int wpb  = blockDim.x >> 5;
  const int r16  = lane & 15;
  const int hi   = lane >> 4;                  // 0 or 1
  const int mTile = blockIdx.y * 16;
  const int nBase = (blockIdx.x * wpb + wave) * (TN * 16);

  const _Float16* aptr = A + (long)(mTile + r16) * lda + (hi << 4);
  // B 32x16 f16 as loaded here (same k order as A): lane = column, 16 contiguous K per lane (hi half -> K+16)
  const _Float16* bptr = Bm + (long)(nBase + r16) * ldb + (hi << 4);
  const long bTileStride = (long)16 * ldb;

  v8f acc[TN];
  v8f zero = {};
#pragma unroll
  for (int t = 0; t < TN; ++t) acc[t] = zero;

  for (int k0 = 0; k0 < K; k0 += 32) {
    v16h a = *(const v16h*)(aptr + k0);
#pragma unroll
    for (int t = 0; t < TN; ++t) {
      v16h b = *(const v16h*)(bptr + t * bTileStride + k0);
      acc[t] = __builtin_amdgcn_wmma_f32_16x16x32_f16(
          false, a, false, b, (short)0, acc[t], false, false);
    }
  }

  // C/D layout: lane(0-15) -> col, VGPR r -> row r (lanes 16-31: row 8+r)
  const int mOff = hi * 8;
#pragma unroll
  for (int t = 0; t < TN; ++t) {
    const int n = nBase + t * 16 + r16;
    float bval = 0.0f;
    if (HAS_BIAS) bval = bias[n];
#pragma unroll
    for (int r = 0; r < 8; ++r) {
      const long m = mTile + mOff + r;
      float v = acc[t][r] + bval;
      if (HAS_ADD) v += addend[m * addStride + n];
      Out[m * (long)ldc + n] = v;
    }
  }
}

// ---------------------------------------------------------------------------
// LSTM cell pointwise: gates[64 x 4096] -> c, h (h stored f16 for next GEMMs)
// ---------------------------------------------------------------------------
__global__ void lstm_cell(const float* __restrict__ gates,
                          float* __restrict__ c,
                          _Float16* __restrict__ hcur,
                          _Float16* __restrict__ hseq, int t) {
  int idx = blockIdx.x * blockDim.x + threadIdx.x;
  if (idx >= BB * HH) return;
  int b = idx >> 10;           // HH = 1024
  int j = idx & (HH - 1);
  const float* g = gates + (long)b * GG;
  float ig = 1.0f / (1.0f + expf(-g[j]));
  float fg = 1.0f / (1.0f + expf(-g[j + HH]));
  float gg = tanhf(g[j + 2 * HH]);
  float og = 1.0f / (1.0f + expf(-g[j + 3 * HH]));
  float cn = fg * c[idx] + ig * gg;
  float hn = og * tanhf(cn);
  c[idx] = cn;
  hcur[idx] = (_Float16)hn;
  hseq[((long)b * TT + t) * HH + j] = (_Float16)hn;
}

// ---------------------------------------------------------------------------
// CRF NLL: one block (32 thr) per sequence.  emissions padded to ldc=32.
// ---------------------------------------------------------------------------
__global__ void crf_nll(const float* __restrict__ em,     // [B*T][CP]
                        const int* __restrict__ labels,   // [B][T]
                        const float* __restrict__ startT,
                        const float* __restrict__ endT,
                        const float* __restrict__ trans,  // [C][C]
                        float* __restrict__ out, int Bn, int Tn) {
  __shared__ float sAlpha[NC];
  __shared__ float sTrans[NC * NC];
  __shared__ float sRed[32];
  const int b = blockIdx.x;
  const int j = threadIdx.x;
  const float* emb = em + (long)b * Tn * CP;
  const int*   lab = labels + (long)b * Tn;

  for (int i = j; i < NC * NC; i += 32) sTrans[i] = trans[i];
  if (j < NC) sAlpha[j] = startT[j] + emb[j];
  __syncthreads();

  // tag-path score (serial on lane 0; tiny)
  float score = 0.0f;
  if (j == 0) {
    int tg0 = lab[0]; int t0 = (tg0 < 0) ? tg0 + NC : tg0;   // jax-style wrap
    score = startT[t0] + emb[t0];
    int prev = t0;
    int cnt = (lab[0] != -1) ? 1 : 0;
    for (int t = 1; t < Tn; ++t) {
      int tg = lab[t];
      int tc = (tg < 0) ? tg + NC : tg;
      float m = (tg != -1) ? 1.0f : 0.0f;
      score += (sTrans[prev * NC + tc] + emb[(long)t * CP + tc]) * m;
      prev = tc;
      cnt += (tg != -1) ? 1 : 0;
    }
    int li = cnt - 1; if (li < 0) li = 0;
    int tl = lab[li]; if (tl < 0) tl += NC;
    score += endT[tl];
  }
  __syncthreads();

  // forward algorithm: lane j owns alpha[j]
  for (int t = 1; t < Tn; ++t) {
    bool mt = (lab[t] != -1);
    float nxt = 0.0f;
    if (j < NC) {
      float mx = -__builtin_inff();
      for (int i = 0; i < NC; ++i) {
        float v = sAlpha[i] + sTrans[i * NC + j];
        mx = fmaxf(mx, v);
      }
      float s = 0.0f;
      for (int i = 0; i < NC; ++i)
        s += expf(sAlpha[i] + sTrans[i * NC + j] - mx);
      nxt = mx + logf(s) + emb[(long)t * CP + j];
    }
    __syncthreads();
    if (j < NC && mt) sAlpha[j] = nxt;
    __syncthreads();
  }

  sRed[j] = (j < NC) ? sAlpha[j] + endT[j] : -__builtin_inff();
  __syncthreads();
  if (j == 0) {
    float mx = -__builtin_inff();
    for (int i = 0; i < NC; ++i) mx = fmaxf(mx, sRed[i]);
    float s = 0.0f;
    for (int i = 0; i < NC; ++i) s += expf(sRed[i] - mx);
    float logZ = mx + logf(s);
    atomicAdd(out, -(score - logZ) / (float)Bn);
  }
}

// ---------------------------------------------------------------------------
extern "C" void kernel_launch(void* const* d_in, const int* in_sizes, int n_in,
                              void* d_out, int out_size, void* d_ws, size_t ws_size,
                              hipStream_t stream) {
  (void)in_sizes; (void)n_in; (void)out_size; (void)ws_size;
  const float* x    = (const float*)d_in[0];
  const int*   lab  = (const int*)d_in[1];
  // d_in[2] lengths: unused by reference math (mask derives from labels)
  const float* Wih0 = (const float*)d_in[3];
  const float* Whh0 = (const float*)d_in[4];
  const float* bih0 = (const float*)d_in[5];
  const float* bhh0 = (const float*)d_in[6];
  const float* Wih1 = (const float*)d_in[7];
  const float* Whh1 = (const float*)d_in[8];
  const float* bih1 = (const float*)d_in[9];
  const float* bhh1 = (const float*)d_in[10];
  const float* fcw  = (const float*)d_in[11];
  const float* fcb  = (const float*)d_in[12];
  const float* stT  = (const float*)d_in[13];
  const float* enT  = (const float*)d_in[14];
  const float* trs  = (const float*)d_in[15];
  float* out = (float*)d_out;

  char* ws = (char*)d_ws;
  size_t off = 0;
  auto alloc = [&](size_t bytes) -> void* {
    void* p = ws + off;
    off = (off + bytes + 255) & ~(size_t)255;
    return p;
  };
  float*    xproj = (float*)   alloc((size_t)BB * TT * GG * 4);  // reused both layers
  _Float16* x16   = (_Float16*)alloc((size_t)BB * TT * II * 2);
  _Float16* wih0h = (_Float16*)alloc((size_t)GG * II * 2);
  _Float16* whh0h = (_Float16*)alloc((size_t)GG * HH * 2);
  _Float16* wih1h = (_Float16*)alloc((size_t)GG * HH * 2);
  _Float16* whh1h = (_Float16*)alloc((size_t)GG * HH * 2);
  _Float16* hseq0 = (_Float16*)alloc((size_t)BB * TT * HH * 2);
  _Float16* hseq1 = (_Float16*)alloc((size_t)BB * TT * HH * 2);
  float*    gates = (float*)   alloc((size_t)BB * GG * 4);
  _Float16* hcur  = (_Float16*)alloc((size_t)BB * HH * 2);
  float*    ccur  = (float*)   alloc((size_t)BB * HH * 4);
  float*    bias0 = (float*)   alloc((size_t)GG * 4);
  float*    bias1 = (float*)   alloc((size_t)GG * 4);
  _Float16* fcw16 = (_Float16*)alloc((size_t)CP * HH * 2);
  float*    fcb32 = (float*)   alloc((size_t)CP * 4);
  float*    emis  = (float*)   alloc((size_t)BB * TT * CP * 4);

  const dim3 blk(256);
  // ---- casts / prep ----
  long nx = (long)BB * TT * II;
  cast_f32_to_f16<<<dim3((unsigned)((nx + 255) / 256)), blk, 0, stream>>>(x, x16, nx);
  cast_f32_to_f16<<<dim3((GG * II + 255) / 256), blk, 0, stream>>>(Wih0, wih0h, (long)GG * II);
  cast_f32_to_f16<<<dim3((GG * HH + 255) / 256), blk, 0, stream>>>(Whh0, whh0h, (long)GG * HH);
  cast_f32_to_f16<<<dim3((GG * HH + 255) / 256), blk, 0, stream>>>(Wih1, wih1h, (long)GG * HH);
  cast_f32_to_f16<<<dim3((GG * HH + 255) / 256), blk, 0, stream>>>(Whh1, whh1h, (long)GG * HH);
  add_bias_pair<<<dim3((GG + 255) / 256), blk, 0, stream>>>(bih0, bhh0, bias0, GG);
  add_bias_pair<<<dim3((GG + 255) / 256), blk, 0, stream>>>(bih1, bhh1, bias1, GG);
  pad_fc<<<dim3((CP * HH + 255) / 256), blk, 0, stream>>>(fcw, fcb, fcw16, fcb32);

  const int M = BB * TT;                       // 32768
  // big projection GEMMs: 256 thr = 8 waves * (TN=4 -> 64 cols) = 512 cols/block
  const dim3 gBig(GG / 512, M / 16);           // (8, 2048)
  // step GEMM: 128 thr = 4 waves * 64 cols = 256 cols/block, M=64 -> 4 row blocks
  const dim3 gStep(GG / 256, BB / 16);         // (16, 4) = 64 blocks
  const dim3 blkStep(128);
  // emissions: TN=2 -> 32 cols per wave, 1 wave per block
  const dim3 gEm(1, M / 16);                   // (1, 2048)
  const dim3 blkEm(32);

  // ---- layer 0: x_proj = x * Wih0^T + (b_ih0 + b_hh0) ----
  gemm_f16_wmma<4, true, false><<<gBig, blk, 0, stream>>>(
      x16, II, wih0h, II, bias0, nullptr, 0, xproj, GG, II);
  zero_state<<<dim3((BB * HH + 255) / 256), blk, 0, stream>>>(hcur, ccur, BB * HH);
  for (int t = 0; t < TT; ++t) {
    gemm_f16_wmma<4, false, true><<<gStep, blkStep, 0, stream>>>(
        hcur, HH, whh0h, HH, nullptr, xproj + (long)t * GG, (long)TT * GG,
        gates, GG, HH);
    lstm_cell<<<dim3((BB * HH + 255) / 256), blk, 0, stream>>>(gates, ccur, hcur, hseq0, t);
  }

  // ---- layer 1 ----
  gemm_f16_wmma<4, true, false><<<gBig, blk, 0, stream>>>(
      hseq0, HH, wih1h, HH, bias1, nullptr, 0, xproj, GG, HH);
  zero_state<<<dim3((BB * HH + 255) / 256), blk, 0, stream>>>(hcur, ccur, BB * HH);
  for (int t = 0; t < TT; ++t) {
    gemm_f16_wmma<4, false, true><<<gStep, blkStep, 0, stream>>>(
        hcur, HH, whh1h, HH, nullptr, xproj + (long)t * GG, (long)TT * GG,
        gates, GG, HH);
    lstm_cell<<<dim3((BB * HH + 255) / 256), blk, 0, stream>>>(gates, ccur, hcur, hseq1, t);
  }

  // ---- emissions = h1 * fc_w^T + fc_b (N padded to 32) ----
  gemm_f16_wmma<2, true, false><<<gEm, blkEm, 0, stream>>>(
      hseq1, HH, fcw16, HH, fcb32, nullptr, 0, emis, CP, HH);

  // ---- CRF NLL mean -> scalar ----
  zero_scalar<<<1, 32, 0, stream>>>(out);
  crf_nll<<<dim3(BB), dim3(32), 0, stream>>>(emis, lab, stT, enT, trs, out, BB, TT);
}
